// GraphConstructor_46514495816134
// MI455X (gfx1250) — compile-verified, matrix operand order checked by reading, not run on hardware
//
#include <hip/hip_runtime.h>
#include <hip/hip_bf16.h>

#define N_TOTAL 8192
#define D_DIM   64
#define K_NB    10
#define K1      11            // top-(K+1); slot 0 = self sentinel
#define THRESH  0.5f
#define NEDGE   (N_TOTAL * K_NB + N_TOTAL)   // 90112
#define ROWS_PER_BLOCK  32
#define WAVES_PER_BLOCK 4
#define COLS_PER_WAVE   (N_TOTAL / WAVES_PER_BLOCK)  // 2048
#define NEGINF (-3.0e38f)

typedef __attribute__((ext_vector_type(16))) __bf16 v16bf;
typedef __attribute__((ext_vector_type(8)))  float  v8f;

union FragU { uint4 q[2]; v16bf v; };

// A/B fragment loader for V_WMMA_F32_16X16X32_BF16.
// 16-bit operand layout (ISA 7.12.2): lanes 0-15 hold M(or N)=lane with
// K = {kh*32+0..7, kh*32+16..23}; lanes 16-31 hold the +8 shifted K chunks.
// Each K-chunk of 8 bf16 is 16 contiguous bytes in row-major memory.
__device__ __forceinline__ v16bf load_frag(const unsigned short* __restrict__ xb,
                                           int rowBase, int kHalf, int lane) {
  int r = rowBase + (lane & 15);
  int k = kHalf * 32 + 8 * (lane >> 4);
  const uint4* p = reinterpret_cast<const uint4*>(xb + r * D_DIM + k);
  FragU u;
  u.q[0] = p[0];   // K = k .. k+7
  u.q[1] = p[2];   // K = k+16 .. k+23
  return u.v;
}

__device__ __forceinline__ unsigned short f2bf(float f) {
  unsigned int u = __float_as_uint(f);
  u = (u + 0x7FFFu + ((u >> 16) & 1u)) >> 16;   // round-to-nearest-even
  return (unsigned short)u;
}

// Unrolled sorted-descending insert into a register-resident (val,idx) list.
__device__ __forceinline__ void topk_insert(float (&tv)[K1], int (&ti)[K1],
                                            float cv, int ci) {
  if (cv <= tv[K1 - 1]) return;
#pragma unroll
  for (int t = 0; t < K1; ++t) {
    bool gt = cv > tv[t];
    float nv = gt ? cv : tv[t];  int ni = gt ? ci : ti[t];
    float ov = gt ? tv[t] : cv;  int oi = gt ? ti[t] : ci;
    tv[t] = nv; ti[t] = ni; cv = ov; ci = oi;
  }
}

// Rows are pre-normalized (x / ||x||) before bf16 conversion, so the WMMA
// output is directly the cosine similarity (eps=1e-8 is ~1e-9 relative,
// far below bf16 quantization error).
__global__ void __launch_bounds__(256) prep_kernel(const float* __restrict__ x,
                                                   unsigned short* __restrict__ xb) {
  int row = blockIdx.x * blockDim.x + threadIdx.x;
  if (row >= N_TOTAL) return;
  const float* xr = x + row * D_DIM;
  unsigned short* o = xb + row * D_DIM;

  float4 v[D_DIM / 4];
  float s = 0.0f;
#pragma unroll
  for (int i = 0; i < D_DIM / 4; ++i) {
    v[i] = reinterpret_cast<const float4*>(xr)[i];
    s += v[i].x * v[i].x + v[i].y * v[i].y + v[i].z * v[i].z + v[i].w * v[i].w;
  }
  float inv = 1.0f / sqrtf(s);
#pragma unroll
  for (int i = 0; i < D_DIM / 4; ++i) {
    uint2 packed;
    packed.x = (unsigned int)f2bf(v[i].x * inv) | ((unsigned int)f2bf(v[i].y * inv) << 16);
    packed.y = (unsigned int)f2bf(v[i].z * inv) | ((unsigned int)f2bf(v[i].w * inv) << 16);
    reinterpret_cast<uint2*>(o)[i] = packed;
  }
}

__global__ void __launch_bounds__(128) simtopk_kernel(
    const unsigned short* __restrict__ xb,
    int* __restrict__ eout) {
  int lane = threadIdx.x & 31;
  int wave = threadIdx.x >> 5;
  int hi   = lane >> 4;
  int n0   = lane & 15;
  int rowBase = blockIdx.x * ROWS_PER_BLOCK;
  int myRow   = rowBase + lane;                 // lane <-> row mapping

  // Per-wave 32x16 sim tile; row stride 20 dwords = 80B (16B-aligned rows,
  // write halves hit disjoint bank ranges). Plus per-slice merge lists.
  __shared__ __align__(16) float tile[WAVES_PER_BLOCK][32][20];
  __shared__ float mval[WAVES_PER_BLOCK][32][K1];
  __shared__ int   midx[WAVES_PER_BLOCK][32][K1];

  // A fragments for this block's 32 rows (two 16-row tiles x K halves).
  v16bf a00 = load_frag(xb, rowBase,      0, lane);
  v16bf a01 = load_frag(xb, rowBase,      1, lane);
  v16bf a10 = load_frag(xb, rowBase + 16, 0, lane);
  v16bf a11 = load_frag(xb, rowBase + 16, 1, lane);

  float tv[K1]; int ti[K1];
#pragma unroll
  for (int t = 0; t < K1; ++t) { tv[t] = NEGINF; ti[t] = myRow; }

  int colBegin = wave * COLS_PER_WAVE;
  for (int j0 = colBegin; j0 < colBegin + COLS_PER_WAVE; j0 += 16) {
    // B-fragment chunks at constant offsets from one strength-reduced base:
    // element offset = (j0+n0)*64 + kh*32 + 8*hi + {0,16}  ->  uint4 index
    // (j0+n0)*8 + hi + {0,2} (kh=0) / {4,6} (kh=1).
    const uint4* pb = reinterpret_cast<const uint4*>(xb) + ((j0 + n0) * 8 + hi);
    FragU ub0, ub1;
    ub0.q[0] = pb[0];  ub0.q[1] = pb[2];
    ub1.q[0] = pb[4];  ub1.q[1] = pb[6];
    // Next B tile is +16 rows = +2048B: folds into global_prefetch_b8 ioffset.
    __builtin_prefetch(reinterpret_cast<const char*>(pb) + 2048, 0, 3);

    v8f c = {}; v8f d = {};
    c = __builtin_amdgcn_wmma_f32_16x16x32_bf16(false, a00, false, ub0.v, (short)0, c, false, false);
    c = __builtin_amdgcn_wmma_f32_16x16x32_bf16(false, a01, false, ub1.v, (short)0, c, false, false);
    d = __builtin_amdgcn_wmma_f32_16x16x32_bf16(false, a10, false, ub0.v, (short)0, d, false, false);
    d = __builtin_amdgcn_wmma_f32_16x16x32_bf16(false, a11, false, ub1.v, (short)0, d, false, false);

    // No diagonal masking here: the self column is filtered at merge time.
#pragma unroll
    for (int p = 0; p < 8; ++p) {
      int rr = p + 8 * hi;
      tile[wave][rr][n0]      = c[p];
      tile[wave][16 + rr][n0] = d[p];
    }
    // Wave-local LDS round trip: ds ops from one wave complete in order,
    // so a dscnt wait suffices (no block barrier needed).
    asm volatile("s_wait_dscnt 0x0" ::: "memory");

    // Each lane scans its own row's 16 fresh sims: one gate for the tile.
    const float4* rowp = reinterpret_cast<const float4*>(&tile[wave][lane][0]);
    float4 q0 = rowp[0], q1 = rowp[1], q2 = rowp[2], q3 = rowp[3];
    float m01 = fmaxf(fmaxf(q0.x, q0.y), fmaxf(q0.z, q0.w));
    float m23 = fmaxf(fmaxf(q1.x, q1.y), fmaxf(q1.z, q1.w));
    float m45 = fmaxf(fmaxf(q2.x, q2.y), fmaxf(q2.z, q2.w));
    float m67 = fmaxf(fmaxf(q3.x, q3.y), fmaxf(q3.z, q3.w));
    float m = fmaxf(fmaxf(m01, m23), fmaxf(m45, m67));
    if (m > tv[K1 - 1]) {
      // Rare slow path (~100x per 2048 cols): re-read from LDS, keep it
      // a real loop so the static code stays small.
#pragma unroll 1
      for (int q = 0; q < 16; ++q)
        topk_insert(tv, ti, tile[wave][lane][q], j0 + q);
    }
  }

  // Merge the 4 column-slice lists per row (one real block barrier).
#pragma unroll
  for (int t = 0; t < K1; ++t) { mval[wave][lane][t] = tv[t]; midx[wave][lane][t] = ti[t]; }
  __syncthreads();

  if (wave == 0) {
    float fv[K1]; int fi[K1];
#pragma unroll
    for (int t = 0; t < K1; ++t) { fv[t] = NEGINF; fi[t] = myRow; }
    topk_insert(fv, fi, 2.0f, myRow);   // self sentinel -> slot 0
#pragma unroll 1
    for (int s = 0; s < WAVES_PER_BLOCK; ++s) {
#pragma unroll 1
      for (int t = 0; t < K1; ++t) {
        float w = mval[s][lane][t];
        int  ix = midx[s][lane][t];
        if (ix != myRow)                // drop the raw self entry + padding
          topk_insert(fv, fi, w, ix);
      }
    }

    // Emit edges in the reference's flat layout:
    // [0,NEDGE) = edge_src, [NEDGE,2*NEDGE) = edge_dst, then f32 edge_weight.
    float* ew = reinterpret_cast<float*>(eout) + 2 * NEDGE;
    int i = myRow;
#pragma unroll
    for (int t = 1; t < K1; ++t) {          // skip slot 0 (self)
      int e = i * K_NB + (t - 1);
      float w = fv[t];
      bool mk = w > THRESH;
      eout[e]          = i;
      eout[NEDGE + e]  = mk ? fi[t] : i;
      ew[e]            = mk ? w : 0.0f;
    }
    int le = N_TOTAL * K_NB + i;            // self-loop edge
    eout[le]         = i;
    eout[NEDGE + le] = i;
    ew[le]           = 1.0f;
  }
}

extern "C" void kernel_launch(void* const* d_in, const int* in_sizes, int n_in,
                              void* d_out, int out_size, void* d_ws, size_t ws_size,
                              hipStream_t stream) {
  const float* x = (const float*)d_in[0];
  unsigned short* xb = (unsigned short*)d_ws;   // 8192*64 bf16 = 1 MB, row-normalized

  prep_kernel<<<N_TOTAL / 256, 256, 0, stream>>>(x, xb);
  simtopk_kernel<<<N_TOTAL / ROWS_PER_BLOCK, 128, 0, stream>>>(xb, (int*)d_out);
}
